// ManualMoELayer_86904368268078
// MI455X (gfx1250) — compile-verified, matrix operand order checked by reading, not run on hardware
//
#include <hip/hip_runtime.h>
#include <hip/hip_bf16.h>
#include <stdint.h>

#define NTOK 8192
#define CDIM 1024
#define NE   8
#define FF   2048
#define FH   1024          // FF half staged in LDS at a time
#define TILE_M 32

typedef __attribute__((ext_vector_type(16))) __bf16 v16bf;
typedef __attribute__((ext_vector_type(8)))  __bf16 v8bf;
typedef __attribute__((ext_vector_type(8)))  float  v8f;
typedef __attribute__((ext_vector_type(4)))  float  v4f;
typedef __attribute__((ext_vector_type(4)))  __bf16 v4bf;
typedef __attribute__((ext_vector_type(4)))  unsigned int v4u;
typedef __attribute__((ext_vector_type(4)))  int v4i;
typedef __attribute__((ext_vector_type(8)))  int v8i;

// ---------------- fp32 -> bf16 conversion (x, w1, w2) ----------------
__global__ void cvt_bf16(const float* __restrict__ s, __bf16* __restrict__ d, int n4) {
    int i = blockIdx.x * blockDim.x + threadIdx.x;
    if (i >= n4) return;
    v4f v = ((const v4f*)s)[i];
    v4bf o;
    o.x = (__bf16)v.x; o.y = (__bf16)v.y; o.z = (__bf16)v.z; o.w = (__bf16)v.w;
    ((v4bf*)d)[i] = o;
}

// ---------------- gating: wave32 per token, top-2 + softmax ----------------
__global__ void gate_kernel(const float* __restrict__ x, const float* __restrict__ gw,
                            int* __restrict__ counts, int* __restrict__ btok,
                            float* __restrict__ bprob) {
    const int lane = threadIdx.x & 31;
    const int wave = threadIdx.x >> 5;
    const int token = blockIdx.x * 8 + wave;
    const float* xr = x + (size_t)token * CDIM;

    float acc[NE];
#pragma unroll
    for (int e = 0; e < NE; ++e) acc[e] = 0.f;
    for (int c = lane; c < CDIM; c += 32) {
        float xv = xr[c];
#pragma unroll
        for (int e = 0; e < NE; ++e) acc[e] += xv * gw[e * CDIM + c];
    }
#pragma unroll
    for (int off = 16; off > 0; off >>= 1) {
#pragma unroll
        for (int e = 0; e < NE; ++e) acc[e] += __shfl_down(acc[e], off, 32);
    }
    if (lane == 0) {
        int i0 = 0; float s0 = acc[0];
#pragma unroll
        for (int e = 1; e < NE; ++e) { if (acc[e] > s0) { s0 = acc[e]; i0 = e; } }
        int i1 = -1; float s1 = -3.4e38f;
#pragma unroll
        for (int e = 0; e < NE; ++e) { if (e != i0 && acc[e] > s1) { s1 = acc[e]; i1 = e; } }
        float e1 = __expf(s1 - s0);
        float inv = 1.f / (1.f + e1);
        float p0 = inv, p1 = e1 * inv;
        int pos = atomicAdd(&counts[i0], 1);
        btok[i0 * NTOK + pos] = token; bprob[i0 * NTOK + pos] = p0;
        pos = atomicAdd(&counts[i1], 1);
        btok[i1 * NTOK + pos] = token; bprob[i1 * NTOK + pos] = p1;
    }
}

// -------- WMMA fragment build: 8-element K chunks per the CDNA5 layout --------
__device__ __forceinline__ v16bf frag(const __bf16* p, int kc) {
    v8bf x0 = *(const v8bf*)(p + kc);
    v8bf x1 = *(const v8bf*)(p + kc + 16);
    return __builtin_shufflevector(x0, x1, 0,1,2,3,4,5,6,7,8,9,10,11,12,13,14,15);
}

// -------- TDM gather: one descriptor pulls up to 16 token rows into LDS --------
__device__ __forceinline__ void tdm_gather16(const __bf16* xb, unsigned int ldsOff,
                                             const int* Ts, int base, int nvalid) {
    int iw[8];
#pragma unroll
    for (int i = 0; i < 8; ++i) {
        int t0 = Ts[base + 2 * i];
        int t1 = Ts[base + 2 * i + 1];
        iw[i] = __builtin_amdgcn_readfirstlane((t1 << 16) | (t0 & 0xFFFF));
    }
    uint64_t ga = (uint64_t)(uintptr_t)xb;
    v4u g0;
    g0.x = (unsigned)__builtin_amdgcn_readfirstlane((int)(1u | (1u << 31)));   // count=1, gather_mode=1, 16b idx
    g0.y = (unsigned)__builtin_amdgcn_readfirstlane((int)ldsOff);              // lds_addr
    g0.z = (unsigned)__builtin_amdgcn_readfirstlane((int)(uint32_t)ga);        // global_addr[31:0]
    g0.w = (unsigned)__builtin_amdgcn_readfirstlane((int)((uint32_t)(ga >> 32) | (2u << 30))); // addr[56:32]|type=2
    v8i g1;
    g1[0] = __builtin_amdgcn_readfirstlane(1 << 16);                 // data_size=1 (2 bytes)
    g1[1] = __builtin_amdgcn_readfirstlane((int)(CDIM << 16));       // tensor_dim0[15:0] @ bit48
    g1[2] = __builtin_amdgcn_readfirstlane((int)((NTOK & 0xFFFF) << 16)); // tensor_dim1[15:0] @ bit80
    g1[3] = __builtin_amdgcn_readfirstlane((int)((NTOK >> 16) | (CDIM << 16))); // dim1 hi | tile_dim0
    g1[4] = __builtin_amdgcn_readfirstlane(nvalid & 0xFFFF);         // tile_dim1 = #valid indices
    g1[5] = __builtin_amdgcn_readfirstlane((int)CDIM);               // tensor_dim0_stride
    g1[6] = 0;
    g1[7] = 0;
    v4i g2, g3;
    g2[0] = iw[0]; g2[1] = iw[1]; g2[2] = iw[2]; g2[3] = iw[3];
    g3[0] = iw[4]; g3[1] = iw[5]; g3[2] = iw[6]; g3[3] = iw[7];
#if defined(__clang_major__) && (__clang_major__ >= 23)
    v8i g4 = {0, 0, 0, 0, 0, 0, 0, 0};
    __builtin_amdgcn_tensor_load_to_lds(g0, g1, g2, g3, g4, 0);
#else
    __builtin_amdgcn_tensor_load_to_lds(g0, g1, g2, g3, 0);
#endif
}

// ---------------- per-expert FFN tile: 32 tokens, WMMA bf16 ----------------
// wave -> (m-half, quad): each wave owns ONE 16-row half and a 256-wide f/n range,
// so only one 256-VGPR A set is live at a time (no scratch spills), while the
// B tiles are shared by the mh=0/1 wave pair through the WGP cache.
__global__ __launch_bounds__(256, 1) void moe_ffn(
    const __bf16* __restrict__ xb, const __bf16* __restrict__ w1b,
    const __bf16* __restrict__ w2b, const int* __restrict__ counts,
    const int* __restrict__ btok, const float* __restrict__ bprob,
    float* __restrict__ out) {

    __shared__ __bf16 Xs[TILE_M][CDIM];   // 64 KB
    __shared__ __bf16 Hs[TILE_M][FH];     // 64 KB
    __shared__ int    Ts[TILE_M];
    __shared__ float  Ps[TILE_M];

    const int e = blockIdx.y;
    const int tile = blockIdx.x;
    const int cnt = counts[e];
    if (tile * TILE_M >= cnt) return;
    const int rem = cnt - tile * TILE_M;
    const int rows = rem < TILE_M ? rem : TILE_M;

    const int tid = threadIdx.x;
    const int lane = tid & 31;
    const int wave = tid >> 5;
    const int half = lane >> 4;       // 0: lanes 0-15, 1: lanes 16-31
    const int l15 = lane & 15;
    const int mh = wave >> 2;         // which 16-row half this wave owns
    const int quad = wave & 3;        // which 256-wide f/n range

    if (tid < TILE_M) {
        if (tid < rows) {
            Ts[tid] = btok[e * NTOK + tile * TILE_M + tid];
            Ps[tid] = bprob[e * NTOK + tile * TILE_M + tid];
        } else { Ts[tid] = 0; Ps[tid] = 0.f; }
    }
    __syncthreads();

    // zero pad rows (only runs on tail tiles)
    for (int r = rows; r < TILE_M; ++r)
        for (int c = tid; c < CDIM; c += 256) Xs[r][c] = (__bf16)0.f;

    // gather token rows: full 16-row groups via TDM gather, tails via async LDS loads
    if (wave < 2) {
        const int b0 = wave * 16;
        int h = rows - b0;
        h = h < 0 ? 0 : (h > 16 ? 16 : h);
        if (h == 16) {
            tdm_gather16(xb, (unsigned int)(uintptr_t)&Xs[b0][0], Ts, b0, 16);
            __builtin_amdgcn_s_wait_tensorcnt(0);
        } else if (h > 0) {
            const int total = h * 128;               // 16B chunks
            for (int idx = lane; idx < total; idx += 32) {
                int r = b0 + (idx >> 7);
                int ch = idx & 127;
                const __bf16* src = xb + (size_t)Ts[r] * CDIM + ch * 8;
                unsigned int ldsOff = (unsigned int)(uintptr_t)(&Xs[r][ch * 8]);
                uint64_t gaddr = (uint64_t)(uintptr_t)src;
                asm volatile("global_load_async_to_lds_b128 %0, %1, off"
                             :: "v"(ldsOff), "v"(gaddr) : "memory");
            }
            asm volatile("s_wait_asynccnt 0" ::: "memory");
        }
    }
    __syncthreads();

    // persistent phase-2 accumulators: 16 n-tiles per wave (128 VGPRs)
    v8f yacc[16];
#pragma unroll
    for (int j = 0; j < 16; ++j) yacc[j] = (v8f){};

    const __bf16* w1e = w1b + (size_t)e * FF * CDIM;
    const __bf16* w2e = w2b + (size_t)e * CDIM * FF;

    for (int ph = 0; ph < 2; ++ph) {
        // ---- phase 1: Hs[mh rows, quad f-range] = silu(X @ W1[ph half]^T) ----
        for (int j = 0; j < 16; ++j) {
            const int fl = quad * 256 + j * 16;        // local f within this half
            const int f = ph * FH + fl;
            v8f acc = {};
            const __bf16* ap = &Xs[mh * 16 + l15][0] + half * 8;
            const __bf16* bp = w1e + (size_t)(f + l15) * CDIM + half * 8;
            for (int kc = 0; kc < CDIM; kc += 32) {
                v16bf b = frag(bp, kc);
                v16bf a = frag(ap, kc);
                acc = __builtin_amdgcn_wmma_f32_16x16x32_bf16(false, a, false, b,
                                                              (short)0, acc, false, false);
            }
#pragma unroll
            for (int r = 0; r < 8; ++r) {
                int m = r + half * 8;
                float v = acc[r];
                Hs[mh * 16 + m][fl + l15] = (__bf16)(v / (1.f + __expf(-v)));  // silu
            }
        }
        __syncthreads();

        // ---- phase 2 partial: yacc += H_half @ W2[:, ph half]^T ----
#pragma unroll
        for (int j = 0; j < 16; ++j) {
            const int n0 = quad * 256 + j * 16;
            v8f acc = yacc[j];
            const __bf16* ap = &Hs[mh * 16 + l15][0] + half * 8;
            const __bf16* bp = w2e + (size_t)(n0 + l15) * FF + ph * FH + half * 8;
            for (int kc = 0; kc < FH; kc += 32) {
                v16bf b = frag(bp, kc);
                v16bf a = frag(ap, kc);
                acc = __builtin_amdgcn_wmma_f32_16x16x32_bf16(false, a, false, b,
                                                              (short)0, acc, false, false);
            }
            yacc[j] = acc;
        }
        __syncthreads();   // Hs reused next half
    }

    // ---- writeout: scale by routing prob, combine across experts with f32 atomics ----
#pragma unroll
    for (int j = 0; j < 16; ++j) {
        const int n0 = quad * 256 + j * 16;
#pragma unroll
        for (int r = 0; r < 8; ++r) {
            int m = mh * 16 + r + half * 8;
            if (m < rows)
                atomicAdd(&out[(size_t)Ts[m] * CDIM + n0 + l15], yacc[j][r] * Ps[m]);
        }
    }
}

extern "C" void kernel_launch(void* const* d_in, const int* in_sizes, int n_in,
                              void* d_out, int out_size, void* d_ws, size_t ws_size,
                              hipStream_t stream) {
    const float* x  = (const float*)d_in[0];
    const float* gw = (const float*)d_in[1];
    const float* w1 = (const float*)d_in[2];
    const float* w2 = (const float*)d_in[3];
    float* out = (float*)d_out;

    char* ws = (char*)d_ws;
    __bf16* xb  = (__bf16*)ws;  ws += (size_t)NTOK * CDIM * 2;
    __bf16* w1b = (__bf16*)ws;  ws += (size_t)NE * FF * CDIM * 2;
    __bf16* w2b = (__bf16*)ws;  ws += (size_t)NE * CDIM * FF * 2;
    int*   counts = (int*)ws;   ws += 256;
    int*   btok   = (int*)ws;   ws += (size_t)NE * NTOK * 4;
    float* bprob  = (float*)ws;

    hipMemsetAsync(counts, 0, 256, stream);
    hipMemsetAsync(d_out, 0, (size_t)NTOK * CDIM * 4, stream);

    {
        int n4 = NTOK * CDIM / 4;
        cvt_bf16<<<(n4 + 255) / 256, 256, 0, stream>>>(x, xb, n4);
    }
    {
        int n4 = NE * FF * CDIM / 4;
        cvt_bf16<<<(n4 + 255) / 256, 256, 0, stream>>>(w1, w1b, n4);
        cvt_bf16<<<(n4 + 255) / 256, 256, 0, stream>>>(w2, w2b, n4);
    }
    gate_kernel<<<NTOK / 8, 256, 0, stream>>>(x, gw, counts, btok, bprob);

    dim3 grid(NTOK / TILE_M, NE);
    moe_ffn<<<grid, 256, 0, stream>>>(xb, w1b, w2b, counts, btok, bprob, out);
}
